// LlamaAttention_44590350467458
// MI455X (gfx1250) — compile-verified, matrix operand order checked by reading, not run on hardware
//
#include <hip/hip_runtime.h>
#include <hip/hip_bf16.h>

#define B_REQ 4
#define SEQL  1024
#define HID   4096
#define NH    32
#define DHD   128
#define RANK  16
#define TTOK  (B_REQ * SEQL)

typedef __attribute__((ext_vector_type(16))) __bf16 bf16x16;
typedef __attribute__((ext_vector_type(8)))  __bf16 bf16x8;
typedef __attribute__((ext_vector_type(8)))  float  f32x8;

static __device__ __forceinline__ bf16x16 bf_combine(bf16x8 lo, bf16x8 hi) {
  return __builtin_shufflevector(lo, hi, 0,1,2,3,4,5,6,7,8,9,10,11,12,13,14,15);
}
static __device__ __forceinline__ f32x8 wmma_bf16(bf16x16 a, bf16x16 b, f32x8 c) {
  return __builtin_amdgcn_wmma_f32_16x16x32_bf16(false, a, false, b, (short)0, c,
                                                 false, false);
}

// ===========================================================================
// FAST GEMM: full tiles only (M%128==0, N%128==0, K%64==0). No guards.
// BK=64, register double-buffered global->LDS staging, 16 WMMAs per stage.
// C[M,N] = A[M,K]*B[K,N] (+C if ACC), f32 in/out, bf16 WMMA compute.
// ===========================================================================
#define FLDS 72   // halves; 144B row stride keeps all b128/b64 LDS ops aligned

template <int ACC>
__global__ __launch_bounds__(256)
void gemm_fast(const float* __restrict__ A, const float* __restrict__ B,
               float* __restrict__ C, int M, int N, int K,
               long long sA, long long sB, long long sC)
{
  __shared__ __bf16 As[128 * FLDS];
  __shared__ __bf16 Bs[128 * FLDS];

  A += (long long)blockIdx.z * sA;
  B += (long long)blockIdx.z * sB;
  C += (long long)blockIdx.z * sC;

  const int tN   = blockIdx.x * 128;
  const int tM   = blockIdx.y * 128;
  const int tid  = threadIdx.x;
  const int lane = tid & 31;
  const int wave = tid >> 5;
  const int wM   = wave & 3;        // 4 waves x 32 rows
  const int wN   = wave >> 2;       // 2 waves x 64 cols
  const int l16  = lane & 15;
  const int half = lane >> 4;
  const int kb   = half * 8;

  f32x8 acc[2][4];
#pragma unroll
  for (int mi = 0; mi < 2; ++mi)
#pragma unroll
    for (int ni = 0; ni < 4; ++ni) acc[mi][ni] = (f32x8){0,0,0,0,0,0,0,0};

  float4 ra[8], rb[8];

  auto loadA = [&](int k0) {
#pragma unroll
    for (int p = 0; p < 8; ++p) {
      int idx = (p * 256 + tid) * 4;
      int r = idx >> 6, c = idx & 63;                 // 128 x 64 tile
      ra[p] = *(const float4*)(A + (long long)(tM + r) * K + k0 + c);
    }
  };
  auto loadB = [&](int k0) {
#pragma unroll
    for (int p = 0; p < 8; ++p) {
      int idx = (p * 256 + tid) * 4;
      int kr = idx >> 7, nc = idx & 127;              // 64 x 128 tile
      rb[p] = *(const float4*)(B + (long long)(k0 + kr) * N + tN + nc);
    }
  };
  auto stage = [&]() {
#pragma unroll
    for (int p = 0; p < 8; ++p) {
      int idx = (p * 256 + tid) * 4;
      int r = idx >> 6, c = idx & 63;
      __bf16* d = &As[r * FLDS + c];                  // 8B ds_store
      d[0] = (__bf16)ra[p].x; d[1] = (__bf16)ra[p].y;
      d[2] = (__bf16)ra[p].z; d[3] = (__bf16)ra[p].w;
    }
#pragma unroll
    for (int p = 0; p < 8; ++p) {
      int idx = (p * 256 + tid) * 4;
      int kr = idx >> 7, nc = idx & 127;
      Bs[(nc + 0) * FLDS + kr] = (__bf16)rb[p].x;     // transpose to Bs[n][k]
      Bs[(nc + 1) * FLDS + kr] = (__bf16)rb[p].y;
      Bs[(nc + 2) * FLDS + kr] = (__bf16)rb[p].z;
      Bs[(nc + 3) * FLDS + kr] = (__bf16)rb[p].w;
    }
  };

  const int kTiles = K >> 6;                          // K / 64
  loadA(0);
  loadB(0);
  for (int kt = 0; kt < kTiles; ++kt) {
    stage();
    __syncthreads();
    if (kt + 1 < kTiles) {                            // overlap next-tile HBM
      loadA((kt + 1) * 64);                           // latency with WMMAs
      loadB((kt + 1) * 64);
    }
#pragma unroll
    for (int ks = 0; ks < 2; ++ks) {                  // two K=32 steps
      bf16x16 af[2], bfv[4];
#pragma unroll
      for (int mi = 0; mi < 2; ++mi) {
        const __bf16* p0 = &As[(wM * 32 + mi * 16 + l16) * FLDS + ks * 32 + kb];
        af[mi] = bf_combine(*(const bf16x8*)p0, *(const bf16x8*)(p0 + 16));
      }
#pragma unroll
      for (int ni = 0; ni < 4; ++ni) {
        const __bf16* p0 = &Bs[(wN * 64 + ni * 16 + l16) * FLDS + ks * 32 + kb];
        bfv[ni] = bf_combine(*(const bf16x8*)p0, *(const bf16x8*)(p0 + 16));
      }
#pragma unroll
      for (int mi = 0; mi < 2; ++mi)
#pragma unroll
        for (int ni = 0; ni < 4; ++ni)
          acc[mi][ni] = wmma_bf16(af[mi], bfv[ni], acc[mi][ni]);
    }
    __syncthreads();
  }

#pragma unroll
  for (int mi = 0; mi < 2; ++mi)
#pragma unroll
    for (int ni = 0; ni < 4; ++ni) {
      int col = tN + wN * 64 + ni * 16 + l16;
#pragma unroll
      for (int i = 0; i < 8; ++i) {
        int row = tM + wM * 32 + mi * 16 + i + half * 8;
        long long off = (long long)row * N + col;
        if (ACC) C[off] += acc[mi][ni][i]; else C[off] = acc[mi][ni][i];
      }
    }
}

// ===========================================================================
// GUARDED GEMM: any M/N/K (used only for rank-16 LoRA shapes). BK=32.
// ===========================================================================
#define LDSA 48

__global__ __launch_bounds__(256)
void gemm_guarded(const float* __restrict__ A, const float* __restrict__ B,
                  float* __restrict__ C, int M, int N, int K,
                  long long sA, long long sB, long long sC, int accumulate)
{
  __shared__ __bf16 As[128 * LDSA];
  __shared__ __bf16 Bs[128 * LDSA];

  A += (long long)blockIdx.z * sA;
  B += (long long)blockIdx.z * sB;
  C += (long long)blockIdx.z * sC;

  const int tN   = blockIdx.x * 128;
  const int tM   = blockIdx.y * 128;
  const int tid  = threadIdx.x;
  const int lane = tid & 31;
  const int wave = tid >> 5;
  const int wM   = wave & 3;
  const int wN   = wave >> 2;
  const int l16  = lane & 15;
  const int half = lane >> 4;
  const int kb   = half * 8;

  f32x8 acc[2][4];
#pragma unroll
  for (int mi = 0; mi < 2; ++mi)
#pragma unroll
    for (int ni = 0; ni < 4; ++ni) acc[mi][ni] = (f32x8){0,0,0,0,0,0,0,0};

  const int kTiles = (K + 31) / 32;
  for (int kt = 0; kt < kTiles; ++kt) {
    const int k0 = kt * 32;
#pragma unroll
    for (int p = 0; p < 4; ++p) {
      int idx = (p * 256 + tid) * 4;
      int r = idx >> 5, c = idx & 31;
      float4 v = make_float4(0.f, 0.f, 0.f, 0.f);
      int gr = tM + r;
      if (gr < M) {
        const float* src = A + (long long)gr * K + k0 + c;
        if (k0 + c + 3 < K) v = *(const float4*)src;
        else {
          if (k0 + c + 0 < K) v.x = src[0];
          if (k0 + c + 1 < K) v.y = src[1];
          if (k0 + c + 2 < K) v.z = src[2];
          if (k0 + c + 3 < K) v.w = src[3];
        }
      }
      __bf16* d = &As[r * LDSA + c];
      d[0] = (__bf16)v.x; d[1] = (__bf16)v.y;
      d[2] = (__bf16)v.z; d[3] = (__bf16)v.w;
    }
#pragma unroll
    for (int p = 0; p < 4; ++p) {
      int idx = (p * 256 + tid) * 4;
      int kr = idx >> 7, nc = idx & 127;
      float4 v = make_float4(0.f, 0.f, 0.f, 0.f);
      if (k0 + kr < K) {
        const float* src = B + (long long)(k0 + kr) * N + tN + nc;
        if (tN + nc + 3 < N) v = *(const float4*)src;
        else {
          if (tN + nc + 0 < N) v.x = src[0];
          if (tN + nc + 1 < N) v.y = src[1];
          if (tN + nc + 2 < N) v.z = src[2];
          if (tN + nc + 3 < N) v.w = src[3];
        }
      }
      Bs[(nc + 0) * LDSA + kr] = (__bf16)v.x;
      Bs[(nc + 1) * LDSA + kr] = (__bf16)v.y;
      Bs[(nc + 2) * LDSA + kr] = (__bf16)v.z;
      Bs[(nc + 3) * LDSA + kr] = (__bf16)v.w;
    }
    __syncthreads();

    bf16x16 af[2], bfv[4];
#pragma unroll
    for (int mi = 0; mi < 2; ++mi) {
      const __bf16* p0 = &As[(wM * 32 + mi * 16 + l16) * LDSA + kb];
      af[mi] = bf_combine(*(const bf16x8*)p0, *(const bf16x8*)(p0 + 16));
    }
#pragma unroll
    for (int ni = 0; ni < 4; ++ni) {
      const __bf16* p0 = &Bs[(wN * 64 + ni * 16 + l16) * LDSA + kb];
      bfv[ni] = bf_combine(*(const bf16x8*)p0, *(const bf16x8*)(p0 + 16));
    }
#pragma unroll
    for (int mi = 0; mi < 2; ++mi)
#pragma unroll
      for (int ni = 0; ni < 4; ++ni)
        acc[mi][ni] = wmma_bf16(af[mi], bfv[ni], acc[mi][ni]);
    __syncthreads();
  }

#pragma unroll
  for (int mi = 0; mi < 2; ++mi)
#pragma unroll
    for (int ni = 0; ni < 4; ++ni) {
      int col = tN + wN * 64 + ni * 16 + l16;
      if (col < N) {
#pragma unroll
        for (int i = 0; i < 8; ++i) {
          int row = tM + wM * 32 + mi * 16 + i + half * 8;
          if (row < M) {
            long long off = (long long)row * N + col;
            float v = acc[mi][ni][i];
            if (accumulate) C[off] += v; else C[off] = v;
          }
        }
      }
    }
}

// ---------------------------------------------------------------------------
// RoPE + f32->bf16 convert (scale folds 1/sqrt(DH) into Q).
// ---------------------------------------------------------------------------
__global__ void rope_cvt_kernel(const float* __restrict__ src,
                                __bf16* __restrict__ dst, float scale)
{
  long long idx = (long long)blockIdx.x * blockDim.x + threadIdx.x;
  int d = (int)(idx & 63);
  long long rest = idx >> 6;
  int h = (int)(rest & (NH - 1));
  long long tok = rest >> 5;
  int s = (int)(tok & (SEQL - 1));
  long long base = tok * (NH * DHD) + h * DHD;
  float x1 = src[base + d];
  float x2 = src[base + d + 64];
  float inv_freq = __expf(-(float)d * (logf(10000.0f) / 64.0f));
  float ang = (float)s * inv_freq;
  float sn, cs;
  __sincosf(ang, &sn, &cs);
  dst[base + d]      = (__bf16)((x1 * cs - x2 * sn) * scale);
  dst[base + d + 64] = (__bf16)((x2 * cs + x1 * sn) * scale);
}

// V [b,s,h,d] f32 -> Vt [b,h,d,s] bf16 (coalesced writes)
__global__ void cvt_vT_kernel(const float* __restrict__ v, __bf16* __restrict__ vT)
{
  long long idx = (long long)blockIdx.x * blockDim.x + threadIdx.x;
  int s = (int)(idx & (SEQL - 1));
  long long r = idx >> 10;
  int d = (int)(r & (DHD - 1)); r >>= 7;
  int h = (int)(r & (NH - 1));
  long long b = r >> 5;
  vT[idx] = (__bf16)(v[((b * SEQL + s) * (NH * DHD)) + h * DHD + d]);
}

// ---------------------------------------------------------------------------
// Flash attention: 8 waves x 16 query rows; online softmax, 32-key tiles.
// ---------------------------------------------------------------------------
__global__ __launch_bounds__(256)
void attn_kernel(const __bf16* __restrict__ Q, const __bf16* __restrict__ Kb,
                 const __bf16* __restrict__ Vt, float* __restrict__ O)
{
  __shared__ __bf16 Plds[8 * 16 * 40];

  const int h    = blockIdx.y;
  const int b    = blockIdx.z;
  const int wave = threadIdx.x >> 5;
  const int lane = threadIdx.x & 31;
  const int l16  = lane & 15;
  const int half = lane >> 4;
  const int kb   = half * 8;
  const int q0   = blockIdx.x * 128 + wave * 16;

  bf16x16 qf[4];
  const __bf16* qbase = Q + ((long long)(b * SEQL + q0 + l16) * (NH * DHD)) + h * DHD;
#pragma unroll
  for (int c = 0; c < 4; ++c)
    qf[c] = bf_combine(*(const bf16x8*)(qbase + c * 32 + kb),
                       *(const bf16x8*)(qbase + c * 32 + kb + 16));

  f32x8 o[8];
  float m[8], l[8];
#pragma unroll
  for (int t = 0; t < 8; ++t) o[t] = (f32x8){0,0,0,0,0,0,0,0};
#pragma unroll
  for (int i = 0; i < 8; ++i) { m[i] = -1e30f; l[i] = 0.f; }

  __bf16* pw = &Plds[wave * 16 * 40];
  const int jend = (q0 + 15) >> 5;

  for (int j = 0; j <= jend; ++j) {
    bf16x16 kf[2][4];
#pragma unroll
    for (int t = 0; t < 2; ++t) {
      const __bf16* kbase =
          Kb + ((long long)(b * SEQL + j * 32 + t * 16 + l16) * (NH * DHD)) + h * DHD;
#pragma unroll
      for (int c = 0; c < 4; ++c)
        kf[t][c] = bf_combine(*(const bf16x8*)(kbase + c * 32 + kb),
                              *(const bf16x8*)(kbase + c * 32 + kb + 16));
    }
    f32x8 S0 = (f32x8){0,0,0,0,0,0,0,0};
    f32x8 S1 = (f32x8){0,0,0,0,0,0,0,0};
#pragma unroll
    for (int c = 0; c < 4; ++c) {
      S0 = wmma_bf16(qf[c], kf[0][c], S0);
      S1 = wmma_bf16(qf[c], kf[1][c], S1);
    }

    const int key0 = j * 32 + l16;
    const int key1 = key0 + 16;
#pragma unroll
    for (int i = 0; i < 8; ++i) {
      const int row = q0 + i + half * 8;
      float s0 = (key0 <= row) ? S0[i] : -1e30f;
      float s1 = (key1 <= row) ? S1[i] : -1e30f;
      float tmax = fmaxf(s0, s1);
#pragma unroll
      for (int off = 1; off < 16; off <<= 1)
        tmax = fmaxf(tmax, __shfl_xor(tmax, off, 16));
      float nm    = fmaxf(m[i], tmax);
      float alpha = __expf(m[i] - nm);
      float p0 = (key0 <= row) ? __expf(s0 - nm) : 0.f;
      float p1 = (key1 <= row) ? __expf(s1 - nm) : 0.f;
      float rs = p0 + p1;
#pragma unroll
      for (int off = 1; off < 16; off <<= 1)
        rs += __shfl_xor(rs, off, 16);
      l[i] = l[i] * alpha + rs;
      m[i] = nm;
#pragma unroll
      for (int t = 0; t < 8; ++t) o[t][i] *= alpha;
      const int rl = i + half * 8;
      pw[rl * 40 + l16]      = (__bf16)p0;
      pw[rl * 40 + 16 + l16] = (__bf16)p1;
    }
    asm volatile("s_wait_dscnt 0" ::: "memory");

    const __bf16* pp = pw + l16 * 40 + kb;
    bf16x16 pf = bf_combine(*(const bf16x8*)pp, *(const bf16x8*)(pp + 16));

    const __bf16* vb = Vt + (((long long)(b * NH + h) * DHD) * SEQL) + (long long)j * 32;
#pragma unroll
    for (int t = 0; t < 8; ++t) {
      const __bf16* vtb = vb + (long long)(t * 16 + l16) * SEQL;
      bf16x16 vf = bf_combine(*(const bf16x8*)(vtb + kb),
                              *(const bf16x8*)(vtb + kb + 16));
      o[t] = wmma_bf16(pf, vf, o[t]);
    }
  }

#pragma unroll
  for (int i = 0; i < 8; ++i) {
    float inv = 1.0f / l[i];
    long long row = (long long)b * SEQL + q0 + i + half * 8;
    float* obase = O + row * (NH * DHD) + h * DHD;
#pragma unroll
    for (int t = 0; t < 8; ++t) obase[t * 16 + l16] = o[t][i] * inv;
  }
}

// ---------------------------------------------------------------------------
extern "C" void kernel_launch(void* const* d_in, const int* in_sizes, int n_in,
                              void* d_out, int out_size, void* d_ws, size_t ws_size,
                              hipStream_t stream)
{
  (void)in_sizes; (void)n_in; (void)out_size; (void)ws_size;
  const float* hs = (const float*)d_in[0];
  const float* Wq = (const float*)d_in[1];
  const float* Wk = (const float*)d_in[2];
  const float* Wv = (const float*)d_in[3];
  const float* Wo = (const float*)d_in[4];
  const float* Aq = (const float*)d_in[5];
  const float* Bq = (const float*)d_in[6];
  const float* Ak = (const float*)d_in[7];
  const float* Bk = (const float*)d_in[8];
  const float* Av = (const float*)d_in[9];
  const float* Bv = (const float*)d_in[10];
  const float* Ao = (const float*)d_in[11];
  const float* Bo = (const float*)d_in[12];
  float* out = (float*)d_out;

  char* ws = (char*)d_ws;
  const size_t f32buf  = (size_t)TTOK * (NH * DHD) * sizeof(float);
  const size_t bf16buf = (size_t)TTOK * (NH * DHD) * sizeof(__bf16);
  float*  qf  = (float*) (ws);
  float*  kf  = (float*) (ws + f32buf);
  float*  vf  = (float*) (ws + 2 * f32buf);
  __bf16* qb  = (__bf16*)(ws + 3 * f32buf);
  __bf16* kbb = (__bf16*)(ws + 3 * f32buf + bf16buf);
  __bf16* vT  = (__bf16*)(ws + 3 * f32buf + 2 * bf16buf);
  float*  u   = (float*) (ws + 3 * f32buf + 3 * bf16buf);
  float*  attn = qf;   // attention output aliases qf (dead after rope)

  auto gemm = [&](const float* A, const float* Bm, float* C, int M, int N, int K,
                  long long sA, long long sB, long long sC, int batch, int acc) {
    dim3 g((N + 127) / 128, (M + 127) / 128, batch);
    if ((M % 128 == 0) && (N % 128 == 0) && (K % 64 == 0)) {
      if (acc) gemm_fast<1><<<g, 256, 0, stream>>>(A, Bm, C, M, N, K, sA, sB, sC);
      else     gemm_fast<0><<<g, 256, 0, stream>>>(A, Bm, C, M, N, K, sA, sB, sC);
    } else {
      gemm_guarded<<<g, 256, 0, stream>>>(A, Bm, C, M, N, K, sA, sB, sC, acc);
    }
  };

  const long long sX  = (long long)SEQL * HID;
  const long long sA1 = (long long)HID * RANK;
  const long long sU  = (long long)SEQL * RANK;
  const long long sB2 = (long long)RANK * (NH * DHD);
  const long long sC2 = (long long)SEQL * (NH * DHD);

  // Q = x*Wq + (x*Aq)*Bq
  gemm(hs, Wq, qf, TTOK, NH * DHD, HID, 0, 0, 0, 1, 0);
  gemm(hs, Aq, u, SEQL, RANK, HID, sX, sA1, sU, B_REQ, 0);
  gemm(u, Bq, qf, SEQL, NH * DHD, RANK, sU, sB2, sC2, B_REQ, 1);
  // K
  gemm(hs, Wk, kf, TTOK, NH * DHD, HID, 0, 0, 0, 1, 0);
  gemm(hs, Ak, u, SEQL, RANK, HID, sX, sA1, sU, B_REQ, 0);
  gemm(u, Bk, kf, SEQL, NH * DHD, RANK, sU, sB2, sC2, B_REQ, 1);
  // V
  gemm(hs, Wv, vf, TTOK, NH * DHD, HID, 0, 0, 0, 1, 0);
  gemm(hs, Av, u, SEQL, RANK, HID, sX, sA1, sU, B_REQ, 0);
  gemm(u, Bv, vf, SEQL, NH * DHD, RANK, sU, sB2, sC2, B_REQ, 1);

  const long long nRope = (long long)TTOK * NH * 64;
  rope_cvt_kernel<<<(unsigned)(nRope / 256), 256, 0, stream>>>(qf, qb,
      0.08838834764831845f);                               // 1/sqrt(128)
  rope_cvt_kernel<<<(unsigned)(nRope / 256), 256, 0, stream>>>(kf, kbb, 1.0f);
  const long long nV = (long long)TTOK * NH * DHD;
  cvt_vT_kernel<<<(unsigned)(nV / 256), 256, 0, stream>>>(vf, vT);

  dim3 ga(SEQL / 128, NH, B_REQ);
  attn_kernel<<<ga, 256, 0, stream>>>(qb, kbb, vT, attn);

  // out = attn*Wo + (attn*Ao)*Bo
  gemm(attn, Wo, out, TTOK, HID, NH * DHD, 0, 0, 0, 1, 0);
  gemm(attn, Ao, u, SEQL, RANK, NH * DHD, sC2, (long long)(NH * DHD) * RANK, sU,
       B_REQ, 0);
  gemm(u, Bo, out, SEQL, HID, RANK, sU, (long long)RANK * HID, sX, B_REQ, 1);
}
